// ComplexMultiHeadAttention_27736898798433
// MI455X (gfx1250) — matrix operand order repaired and checked by reading, not compiled, not run on hardware
//
#include <hip/hip_runtime.h>

// ---------------------------------------------------------------------------
// Complex multi-head attention, CDNA5 (gfx1250), wave32 + WMMA f16->f32.
// Pipeline:
//   1) convert weights f32->f16, split complex inputs into r/i f16 planes
//   2) complex GEMM (WMMA) for Q/K/V projections  -> f16 planes [4096,1024]
//   3) flash-style complex attention (WMMA + TDM tensor_load_to_lds staging)
//   4) complex GEMM (WMMA) output projection -> interleaved f32 [B,S,1024,2]
// ---------------------------------------------------------------------------

typedef _Float16 half8  __attribute__((ext_vector_type(8)));
typedef _Float16 half16 __attribute__((ext_vector_type(16)));
typedef float    f32x8  __attribute__((ext_vector_type(8)));
typedef unsigned int u32x4 __attribute__((ext_vector_type(4)));
typedef int          i32x8 __attribute__((ext_vector_type(8)));
typedef int          i32x4 __attribute__((ext_vector_type(4)));

#define D_MODEL 1024
#define SEQ     1024
#define BATCH   4
#define NHEAD   16
#define DHEAD   64
#define MTOT    (BATCH * SEQ)   // 4096 rows

static __device__ inline half16 cat8(half8 lo, half8 hi) {
  return __builtin_shufflevector(lo, hi, 0,1,2,3,4,5,6,7,8,9,10,11,12,13,14,15);
}

// A fragment (16M x 32K, f16), source row-major [rows, ld] with K contiguous.
// ISA layout: lane m = lane&15, half = lane>>4; v0..3 <- K = half*8+0..7,
// v4..7 <- K = 16+half*8+0..7. Two 16B contiguous loads per lane.
static __device__ inline half16 load_frag_a(const _Float16* p, int ld, int lane) {
  int r = lane & 15, hh = lane >> 4;
  const _Float16* q = p + (size_t)r * (size_t)ld + hh * 8;
  half8 lo = *(const half8*)(q);
  half8 hi = *(const half8*)(q + 16);
  return cat8(lo, hi);
}

// B fragment (32K x 16N, f16), sourced from row-major [n rows, ld] with K
// contiguous (i.e. B^T storage, as with torch Linear weights [N,K]).
// ISA layout: lane n = lane&15 holds column n; v0..3 <- K = half*8+0..7,
// v4..7 <- K = 16+half*8+0..7. Two 16B contiguous loads per lane.
static __device__ inline half16 load_frag_b(const _Float16* p, int ld, int lane) {
  int r = lane & 15, hh = lane >> 4;
  const _Float16* q = p + (size_t)r * (size_t)ld + hh * 8;
  half8 lo = *(const half8*)(q);
  half8 hi = *(const half8*)(q + 16);
  return cat8(lo, hi);
}

static __device__ inline f32x8 wmma_f16(half16 a, half16 b, f32x8 c) {
  // (neg_a, A, neg_b, B, c_mod, C, reuse_a, reuse_b)
  return __builtin_amdgcn_wmma_f32_16x16x32_f16(false, a, false, b, (short)0, c,
                                                false, false);
}

// Issue one TDM 2-D tile load (global -> LDS) of a 32x64 f16 tile with row
// stride D_MODEL. D# bitfields per CDNA5 ISA ch.8 (groups 0/1; 2-D tile so
// the remaining descriptor groups are zero). 6-arg builtin form (clang-23).
static __device__ inline void tdm_load_tile_32x64(unsigned lds_addr,
                                                  const _Float16* gptr) {
  const unsigned long long ga = (unsigned long long)(const void*)gptr;
  u32x4 g0;
  g0[0] = 1u;                                   // count=1, user descriptor
  g0[1] = lds_addr;                             // lds_addr  [63:32]
  g0[2] = (unsigned)(ga & 0xFFFFFFFFull);       // global_addr [95:64]
  g0[3] = (unsigned)((ga >> 32) & 0x01FFFFFFull) | 0x80000000u; // [120:96]|type=2

  i32x8 g1;
  g1[0] = 1 << 16;                              // data_size=1 (2 bytes/elem)
  g1[1] = (D_MODEL & 0xFFFF) << 16;             // tensor_dim0 lo16 @ [63:48]
  g1[2] = (D_MODEL >> 16) | ((MTOT & 0xFFFF) << 16); // td0 hi16, tensor_dim1 lo16
  g1[3] = (MTOT >> 16) | (DHEAD << 16);         // td1 hi16, tile_dim0=64
  g1[4] = 32;                                   // tile_dim1=32, tile_dim2=0
  g1[5] = D_MODEL;                              // tensor_dim0_stride lo32
  g1[6] = 0;                                    // stride hi16, dim1_stride lo16
  g1[7] = 0;

  i32x4 gz4;
  gz4[0] = 0; gz4[1] = 0; gz4[2] = 0; gz4[3] = 0;
  i32x8 gz8;
  gz8[0] = 0; gz8[1] = 0; gz8[2] = 0; gz8[3] = 0;
  gz8[4] = 0; gz8[5] = 0; gz8[6] = 0; gz8[7] = 0;
  __builtin_amdgcn_tensor_load_to_lds(g0, g1, gz4, gz4, gz8, 0 /*cpol*/);
}

// ---------------------------------------------------------------------------
// Conversion kernels
// ---------------------------------------------------------------------------
__global__ __launch_bounds__(256) void f32_to_f16_kernel(const float* __restrict__ in,
                                                         _Float16* __restrict__ out,
                                                         int n) {
  int i = blockIdx.x * 256 + threadIdx.x;
  if (i < n) out[i] = (_Float16)in[i];
}

__global__ __launch_bounds__(256) void split_complex_kernel(const float* __restrict__ in,
                                                            _Float16* __restrict__ re,
                                                            _Float16* __restrict__ im,
                                                            int n) {
  int i = blockIdx.x * 256 + threadIdx.x;
  if (i < n) {
    re[i] = (_Float16)in[2 * i + 0];
    im[i] = (_Float16)in[2 * i + 1];
  }
}

// ---------------------------------------------------------------------------
// Complex GEMM:  C = (Ar + i Ai) @ (Wr + i Wi)^T,  A: [M,K] planes, W: [N,K]
//   Cr = Ar Wr^T - Ai Wi^T ;  Ci = Ai Wr^T + Ar Wi^T
// Block: 256 thr (8 waves), tile 64M x 128N; wave tile 16M x 64N.
// Four WMMA product accumulators per 16x16 tile (f16 WMMA has no A/B negate).
// ---------------------------------------------------------------------------
template <bool OUT_F32_PAIR>
__global__ __launch_bounds__(256) void cgemm_kernel(
    const _Float16* __restrict__ Ar, const _Float16* __restrict__ Ai,
    const _Float16* __restrict__ Wr, const _Float16* __restrict__ Wi,
    _Float16* __restrict__ Cr, _Float16* __restrict__ Ci,
    float* __restrict__ Cpair, int N, int K) {
  const int lane = threadIdx.x & 31;
  const int wid  = threadIdx.x >> 5;
  const int wm = wid & 3;    // 4 waves along M
  const int wn = wid >> 2;   // 2 waves along N
  const int m0 = blockIdx.y * 64 + wm * 16;
  const int n0 = blockIdx.x * 128 + wn * 64;

  f32x8 acc[4][4] = {};  // [product 0..3][n-tile 0..3]

  const _Float16* arp = Ar + (size_t)m0 * K;
  const _Float16* aip = Ai + (size_t)m0 * K;

  for (int k0 = 0; k0 < K; k0 += 32) {
    if (k0 + 32 < K) {  // hint next K-step A rows toward the caches
      __builtin_prefetch(arp + (size_t)(lane & 15) * K + k0 + 32, 0, 1);
      __builtin_prefetch(aip + (size_t)(lane & 15) * K + k0 + 32, 0, 1);
    }
    half16 a_r = load_frag_a(arp + k0, K, lane);
    half16 a_i = load_frag_a(aip + k0, K, lane);
#pragma unroll
    for (int nt = 0; nt < 4; ++nt) {
      const size_t wo = (size_t)(n0 + nt * 16) * K + k0;
      half16 b_r = load_frag_b(Wr + wo, K, lane);
      half16 b_i = load_frag_b(Wi + wo, K, lane);
      acc[0][nt] = wmma_f16(a_r, b_r, acc[0][nt]);
      acc[1][nt] = wmma_f16(a_i, b_i, acc[1][nt]);
      acc[2][nt] = wmma_f16(a_i, b_r, acc[2][nt]);
      acc[3][nt] = wmma_f16(a_r, b_i, acc[3][nt]);
    }
  }

  const int cn = lane & 15, hh = lane >> 4;
#pragma unroll
  for (int nt = 0; nt < 4; ++nt) {
    const int col = n0 + nt * 16 + cn;
#pragma unroll
    for (int v = 0; v < 8; ++v) {
      const int row = m0 + v + 8 * hh;   // 16x16 f32 C/D layout
      const float re = acc[0][nt][v] - acc[1][nt][v];
      const float im = acc[2][nt][v] + acc[3][nt][v];
      if (OUT_F32_PAIR) {
        float2 t; t.x = re; t.y = im;
        reinterpret_cast<float2*>(Cpair)[(size_t)row * N + col] = t;
      } else {
        Cr[(size_t)row * N + col] = (_Float16)re;
        Ci[(size_t)row * N + col] = (_Float16)im;
      }
    }
  }
}

// ---------------------------------------------------------------------------
// Flash-style complex attention.
// Grid: (S/64, B*H); block 128 thr (4 waves); wave owns a 16-row q tile.
// Per 32-key step: K tiles staged via TDM tensor_load_to_lds (TENSORcnt),
// V staged transposed by the waves; scores via WMMA, dual online softmax,
// P round-trip through LDS (C-layout -> A-layout), P@V via WMMA into 4
// running product accumulators; normalization + complex combine at the end.
// ---------------------------------------------------------------------------
__global__ __launch_bounds__(128) void cattn_kernel(
    const _Float16* __restrict__ Qr, const _Float16* __restrict__ Qi,
    const _Float16* __restrict__ Kr, const _Float16* __restrict__ Ki,
    const _Float16* __restrict__ Vr, const _Float16* __restrict__ Vi,
    _Float16* __restrict__ Or_, _Float16* __restrict__ Oi_) {
  __shared__ __align__(16) _Float16 sKr[32][DHEAD];
  __shared__ __align__(16) _Float16 sKi[32][DHEAD];
  __shared__ __align__(16) _Float16 sVr[DHEAD][32];  // transposed: [d][key]
  __shared__ __align__(16) _Float16 sVi[DHEAD][32];
  __shared__ __align__(16) _Float16 sPr[4][16][32];  // per-wave P scratch
  __shared__ __align__(16) _Float16 sPi[4][16][32];

  const int tid = threadIdx.x, lane = tid & 31, wid = tid >> 5;
  const int bh = blockIdx.y;
  const int b = bh >> 4, h = bh & 15;
  const int q0 = blockIdx.x * 64 + wid * 16;  // wave's 16 query rows

  // Preload Q fragments (dh=64 -> 2 K-steps) for r and i planes.
  const size_t qbase = (size_t)(b * SEQ + q0) * D_MODEL + h * DHEAD;
  half16 aqr[2], aqi[2];
#pragma unroll
  for (int ks = 0; ks < 2; ++ks) {
    aqr[ks] = load_frag_a(Qr + qbase + ks * 32, D_MODEL, lane);
    aqi[ks] = load_frag_a(Qi + qbase + ks * 32, D_MODEL, lane);
  }

  f32x8 OA[4] = {}, OB[4] = {}, OC[4] = {}, OD[4] = {};  // PrVr PiVi PrVi PiVr
  float mr[8], lr[8], mi[8], li[8];
#pragma unroll
  for (int v = 0; v < 8; ++v) { mr[v] = -1e30f; mi[v] = -1e30f; lr[v] = 0.f; li[v] = 0.f; }

  const int cn = lane & 15, hh = lane >> 4;
  const unsigned lds_kr = (unsigned)(unsigned long long)(const void*)&sKr[0][0];
  const unsigned lds_ki = (unsigned)(unsigned long long)(const void*)&sKi[0][0];

  for (int kb = 0; kb < SEQ; kb += 32) {
    __syncthreads();  // previous iteration's LDS reads complete

    // --- K tiles: one TDM DMA per plane (global -> LDS), issued by wave 0 ---
    if (wid == 0) {
      const size_t kg = (size_t)(b * SEQ + kb) * D_MODEL + h * DHEAD;
      tdm_load_tile_32x64(lds_kr, Kr + kg);
      tdm_load_tile_32x64(lds_ki, Ki + kg);
    }

    // --- V tiles: staged transposed by the waves (TDM cannot transpose) ---
    {
      const int srow = tid >> 2;           // 0..31 (key within tile)
      const int scc  = (tid & 3) * 16;     // d chunk
      const size_t g = (size_t)(b * SEQ + kb + srow) * D_MODEL + h * DHEAD + scc;
      half8 x0 = *(const half8*)(Vr + g);
      half8 x1 = *((const half8*)(Vr + g) + 1);
#pragma unroll
      for (int j = 0; j < 8; ++j) { sVr[scc + j][srow] = x0[j]; sVr[scc + 8 + j][srow] = x1[j]; }
      x0 = *(const half8*)(Vi + g);
      x1 = *((const half8*)(Vi + g) + 1);
#pragma unroll
      for (int j = 0; j < 8; ++j) { sVi[scc + j][srow] = x0[j]; sVi[scc + 8 + j][srow] = x1[j]; }
    }
    if (wid == 0) __builtin_amdgcn_s_wait_tensorcnt(0);  // TDM tiles landed
    __syncthreads();                                     // publish K and V

    // --- scores: 16 q-rows x 32 keys, real & imag (two 16x16 tiles each) ---
    f32x8 sr[2], si[2];
#pragma unroll
    for (int ns = 0; ns < 2; ++ns) {
      f32x8 p1 = {}, p2 = {}, p3 = {}, p4 = {};
#pragma unroll
      for (int ks = 0; ks < 2; ++ks) {
        half16 bkr = load_frag_b(&sKr[ns * 16][ks * 32], DHEAD, lane);
        half16 bki = load_frag_b(&sKi[ns * 16][ks * 32], DHEAD, lane);
        p1 = wmma_f16(aqr[ks], bkr, p1);
        p2 = wmma_f16(aqi[ks], bki, p2);
        p3 = wmma_f16(aqr[ks], bki, p3);
        p4 = wmma_f16(aqi[ks], bkr, p4);
      }
      sr[ns] = (p1 - p2) * 0.125f;  // 1/sqrt(64)
      si[ns] = (p3 + p4) * 0.125f;
    }

    // --- dual online softmax. Row v+8*hh lives across the 16-lane half-group,
    //     so xor masks 1/2/4/8 implement row-wise reductions. ---
    float fr[8], fi[8];
#pragma unroll
    for (int v = 0; v < 8; ++v) {
      float t = fmaxf(sr[0][v], sr[1][v]);
#pragma unroll
      for (int mk = 1; mk < 16; mk <<= 1) t = fmaxf(t, __shfl_xor(t, mk, 32));
      float nm = fmaxf(mr[v], t);
      fr[v] = __expf(mr[v] - nm);
      mr[v] = nm;
      float e0 = __expf(sr[0][v] - nm), e1 = __expf(sr[1][v] - nm);
      sr[0][v] = e0; sr[1][v] = e1;
      float rs = e0 + e1;
#pragma unroll
      for (int mk = 1; mk < 16; mk <<= 1) rs += __shfl_xor(rs, mk, 32);
      lr[v] = lr[v] * fr[v] + rs;

      t = fmaxf(si[0][v], si[1][v]);
#pragma unroll
      for (int mk = 1; mk < 16; mk <<= 1) t = fmaxf(t, __shfl_xor(t, mk, 32));
      nm = fmaxf(mi[v], t);
      fi[v] = __expf(mi[v] - nm);
      mi[v] = nm;
      e0 = __expf(si[0][v] - nm); e1 = __expf(si[1][v] - nm);
      si[0][v] = e0; si[1][v] = e1;
      rs = e0 + e1;
#pragma unroll
      for (int mk = 1; mk < 16; mk <<= 1) rs += __shfl_xor(rs, mk, 32);
      li[v] = li[v] * fi[v] + rs;
    }

    // --- re-stripe P: C-layout -> LDS -> A-layout ---
#pragma unroll
    for (int ns = 0; ns < 2; ++ns)
#pragma unroll
      for (int v = 0; v < 8; ++v) {
        sPr[wid][v + 8 * hh][ns * 16 + cn] = (_Float16)sr[ns][v];
        sPi[wid][v + 8 * hh][ns * 16 + cn] = (_Float16)si[ns][v];
      }
    __syncthreads();

    // --- rescale running accumulators, then accumulate P @ V ---
#pragma unroll
    for (int nt = 0; nt < 4; ++nt)
#pragma unroll
      for (int v = 0; v < 8; ++v) {
        OA[nt][v] = OA[nt][v] * fr[v];
        OC[nt][v] = OC[nt][v] * fr[v];
        OB[nt][v] = OB[nt][v] * fi[v];
        OD[nt][v] = OD[nt][v] * fi[v];
      }
    half16 apr = load_frag_a(&sPr[wid][0][0], 32, lane);
    half16 api = load_frag_a(&sPi[wid][0][0], 32, lane);
#pragma unroll
    for (int nt = 0; nt < 4; ++nt) {
      half16 bvr = load_frag_b(&sVr[nt * 16][0], 32, lane);
      half16 bvi = load_frag_b(&sVi[nt * 16][0], 32, lane);
      OA[nt] = wmma_f16(apr, bvr, OA[nt]);
      OB[nt] = wmma_f16(api, bvi, OB[nt]);
      OC[nt] = wmma_f16(apr, bvi, OC[nt]);
      OD[nt] = wmma_f16(api, bvr, OD[nt]);
    }
  }

  // --- epilogue: normalize per-row, complex combine, store f16 planes ---
  float ilr[8], ili[8];
#pragma unroll
  for (int v = 0; v < 8; ++v) { ilr[v] = 1.0f / lr[v]; ili[v] = 1.0f / li[v]; }
#pragma unroll
  for (int nt = 0; nt < 4; ++nt)
#pragma unroll
    for (int v = 0; v < 8; ++v) {
      const float re = OA[nt][v] * ilr[v] - OB[nt][v] * ili[v];
      const float im = OC[nt][v] * ilr[v] + OD[nt][v] * ili[v];
      const size_t o = (size_t)(b * SEQ + q0 + v + 8 * hh) * D_MODEL +
                       h * DHEAD + nt * 16 + cn;
      Or_[o] = (_Float16)re;
      Oi_[o] = (_Float16)im;
    }
}

// ---------------------------------------------------------------------------
// Host-side launcher
// ---------------------------------------------------------------------------
extern "C" void kernel_launch(void* const* d_in, const int* in_sizes, int n_in,
                              void* d_out, int out_size, void* d_ws, size_t ws_size,
                              hipStream_t stream) {
  (void)in_sizes; (void)n_in; (void)out_size; (void)ws_size;

  const float* q_in = (const float*)d_in[0];
  const float* k_in = (const float*)d_in[1];
  const float* v_in = (const float*)d_in[2];

  char* ws = (char*)d_ws;
  size_t off = 0;
  auto take = [&](size_t elems) {
    _Float16* p = (_Float16*)(ws + off);
    off += elems * sizeof(_Float16);
    return p;
  };

  const size_t WSZ = (size_t)D_MODEL * D_MODEL;  // 1M weight elements
  const size_t PLZ = (size_t)MTOT * D_MODEL;     // 4M plane elements

  _Float16* w16[8];
  for (int i = 0; i < 8; ++i) w16[i] = take(WSZ);  // wq_r,wq_i,wk_r,wk_i,wv_r,wv_i,wo_r,wo_i
  _Float16* xqr = take(PLZ); _Float16* xqi = take(PLZ);
  _Float16* xkr = take(PLZ); _Float16* xki = take(PLZ);
  _Float16* xvr = take(PLZ); _Float16* xvi = take(PLZ);
  _Float16* Qr = take(PLZ); _Float16* Qi = take(PLZ);
  _Float16* Kr = take(PLZ); _Float16* Ki = take(PLZ);
  _Float16* Vr = take(PLZ); _Float16* Vi = take(PLZ);
  _Float16* Or_ = take(PLZ); _Float16* Oi_ = take(PLZ);

  // 1) precision conversion
  for (int i = 0; i < 8; ++i)
    f32_to_f16_kernel<<<(int)(WSZ / 256), 256, 0, stream>>>(
        (const float*)d_in[3 + i], w16[i], (int)WSZ);
  split_complex_kernel<<<(int)(PLZ / 256), 256, 0, stream>>>(q_in, xqr, xqi, (int)PLZ);
  split_complex_kernel<<<(int)(PLZ / 256), 256, 0, stream>>>(k_in, xkr, xki, (int)PLZ);
  split_complex_kernel<<<(int)(PLZ / 256), 256, 0, stream>>>(v_in, xvr, xvi, (int)PLZ);

  // 2) projections: [4096,1024] x [1024,1024]^T complex
  dim3 gg(D_MODEL / 128, MTOT / 64);
  cgemm_kernel<false><<<gg, 256, 0, stream>>>(xqr, xqi, w16[0], w16[1], Qr, Qi,
                                              nullptr, D_MODEL, D_MODEL);
  cgemm_kernel<false><<<gg, 256, 0, stream>>>(xkr, xki, w16[2], w16[3], Kr, Ki,
                                              nullptr, D_MODEL, D_MODEL);
  cgemm_kernel<false><<<gg, 256, 0, stream>>>(xvr, xvi, w16[4], w16[5], Vr, Vi,
                                              nullptr, D_MODEL, D_MODEL);

  // 3) flash-style complex attention (TDM-staged K tiles)
  cattn_kernel<<<dim3(SEQ / 64, BATCH * NHEAD), 128, 0, stream>>>(
      Qr, Qi, Kr, Ki, Vr, Vi, Or_, Oi_);

  // 4) output projection -> interleaved f32 [B,S,1024,2]
  cgemm_kernel<true><<<gg, 256, 0, stream>>>(Or_, Oi_, w16[6], w16[7], nullptr,
                                             nullptr, (float*)d_out, D_MODEL,
                                             D_MODEL);
}
